// ContTimeLSTMCell_42107859370761
// MI455X (gfx1250) — matrix operand order repaired and checked by reading, not compiled, not run on hardware
//
#include <hip/hip_runtime.h>
#include <stdint.h>
#include <stddef.h>

// Problem constants (match reference)
#define H   512
#define Bb  64
#define T   256
#define G7  (7 * H)     // 3584 fused gate width
#define K2H (2 * H)     // 1024 GEMM K (concat [x, h])
#define KHALF (K2H / 2) // 512 per wave K-slice
#define NBLK (H / 16)   // 32 persistent blocks, one per 16-col group of H
#define NTHR 256        // 8 waves = 2 per SIMD32: (Mtile m, K-half s)
#define AROW 1032       // LDS row stride in bf16 elems (1024 data + 8 pad)

typedef __attribute__((ext_vector_type(16))) __bf16   v16bf;
typedef __attribute__((ext_vector_type(8)))  float    v8f;
typedef __attribute__((ext_vector_type(4)))  unsigned u32x4;
typedef __attribute__((ext_vector_type(4)))  float    f32x4;

__device__ __forceinline__ unsigned short f2bf(float f) {
  union { float f; unsigned u; } c; c.f = f;
  unsigned u = c.u;
  unsigned r = u + 0x7FFFu + ((u >> 16) & 1u);  // round-to-nearest-even
  return (unsigned short)(r >> 16);
}

__device__ __forceinline__ float sigf(float x) {
  return 1.0f / (1.0f + __expf(-x));
}
__device__ __forceinline__ float softplusf(float x) {
  return fmaxf(x, 0.0f) + log1pf(__expf(-fabsf(x)));   // stable softplus
}

// ---------------------------------------------------------------------------
// Kernel 1: zero h_init state and the global step-barrier counter
// ---------------------------------------------------------------------------
__global__ void ctl_init(float* __restrict__ hinit, unsigned* __restrict__ bar) {
  int idx = blockIdx.x * blockDim.x + threadIdx.x;
  if (idx < Bb * H) hinit[idx] = 0.0f;
  if (idx == 0) *bar = 0u;
}

// ---------------------------------------------------------------------------
// Kernel 2: W [2H, 7H] fp32 row-major -> Wt [7H][2H] bf16 (K contiguous per
// output column == per-lane-contiguous B fragment for wmma bf16). One-time
// 14.7MB -> 7.3MB transform; L2-resident thereafter.
// ---------------------------------------------------------------------------
__global__ void ctl_wconvert(const float* __restrict__ W, unsigned short* __restrict__ Wt) {
  int idx = blockIdx.x * blockDim.x + threadIdx.x;  // read-coalesced over W
  if (idx < K2H * G7) {
    int k = idx / G7;
    int o = idx - k * G7;
    Wt[(size_t)o * K2H + k] = f2bf(W[idx]);
  }
}

// ---------------------------------------------------------------------------
// Kernel 3: persistent continuous-time LSTM scan.
//   32 blocks (16-col groups) x 256 threads (8 waves, 2 per SIMD32).
//   wave (m = w&3, s = w>>2): all 7 gates for M-rows [16m,16m+16) over the
//   K-slice [512s, 512s+512). GEMM inner loop is branch-free and identical
//   for every wave; unroll capped at 4 to keep VGPRs < 256 (no spills).
//   s=1 waves hand partial sums to s=0 through 28KB LDS scratch; s=0 adds
//   and runs the cell update. c_func/c_bar persist in s=0 registers in WMMA
//   C layout for all 256 steps.
// ---------------------------------------------------------------------------
__launch_bounds__(NTHR, 1)
__global__ void ctl_scan(const float* __restrict__ seq,        // [B,T,H]
                         const float* __restrict__ dtime,      // [B,T]
                         const unsigned short* __restrict__ Wt,// [7H][2H] bf16
                         const float* __restrict__ bias,       // [7H]
                         const float* __restrict__ hinit,      // [B,H] zeros
                         float* __restrict__ hout,             // [B,T,H]
                         float* __restrict__ dout,             // [B,T,4,H]
                         unsigned* __restrict__ bar) {
  extern __shared__ unsigned short Abuf[];          // [64][AROW] bf16 = 132096 B
  float* scr = (float*)(Abuf + 64 * AROW);          // [4 m][7 g][32 lane][8] = 28672 B

  const int tid  = threadIdx.x;
  const int wav  = tid >> 5;
  const int lane = tid & 31;
  const int m    = wav & 3;    // M-tile
  const int s    = wav >> 2;   // K-half
  const int n    = lane & 15;  // WMMA N / output column in tile
  const int hi   = lane >> 4;  // lane half
  const int cbase = blockIdx.x * 16;
  const int col   = cbase + n;
  const int k0    = s * KHALF; // this wave's K-slice start

  // Time-invariant per-lane constants: 32-bit B-fragment offsets and biases
  int   wtb[7];
  float bk[7];
#pragma unroll
  for (int g = 0; g < 7; ++g) {
    int o = g * H + col;
    wtb[g] = o * K2H + hi * 8 + k0;    // lane: K 8h..8h+7 and 16+8h..16+8h+7 bf16 of column o
    bk[g]  = bias[o];
  }
  const int arow  = m * 16 + n;                     // A row this lane supplies
  const int abase = arow * AROW + hi * 8 + k0;      // bf16 A-fragment offset
  const int sbase = (m * 7) * 32 + lane;            // scratch base (per m, lane)

  v8f c_func = {};  // carried states (only meaningful in s==0 waves)
  v8f c_bar  = {};

  for (int t = 0; t < T; ++t) {
    // ---- Stage A = [x_t | h_{t-1}] as bf16 into LDS (coalesced f32x4 reads)
    for (int c = tid; c < Bb * (H / 4); c += NTHR) {
      int row = c >> 7;          // / (H/4)
      int q   = c & 127;
      f32x4 xv = reinterpret_cast<const f32x4*>(
          seq + (size_t)row * T * H + (size_t)t * H)[q];
      f32x4 hv;
      if (t == 0) {
        hv = reinterpret_cast<const f32x4*>(hinit + row * H)[q];
      } else {
        hv = reinterpret_cast<const f32x4*>(
            hout + (size_t)row * T * H + (size_t)(t - 1) * H)[q];
      }
      unsigned short* ax = Abuf + row * AROW + 4 * q;
      ax[0] = f2bf(xv.x); ax[1] = f2bf(xv.y); ax[2] = f2bf(xv.z); ax[3] = f2bf(xv.w);
      unsigned short* ah = Abuf + row * AROW + H + 4 * q;
      ah[0] = f2bf(hv.x); ah[1] = f2bf(hv.y); ah[2] = f2bf(hv.z); ah[3] = f2bf(hv.w);
      if (t + 1 < T)  // pre-touch next step's x tile (global_prefetch_b8)
        __builtin_prefetch(seq + (size_t)row * T * H + (size_t)(t + 1) * H + 4 * q, 0, 1);
    }
    __syncthreads();

    // ---- 7-gate GEMM over this wave's K-slice: branch-free, uniform.
    //      Unroll capped at 4 (the schedule the compiler picked in the good
    //      round-1 codegen) so VGPR pressure stays < 256 and no spilling.
    v8f acc[7] = {};
#pragma unroll 4
    for (int kk = 0; kk < KHALF; kk += 32) {
      union { u32x4 q[2]; v16bf v; } a;
      a.q[0] = *reinterpret_cast<const u32x4*>(Abuf + abase + kk);       // K+0..7
      a.q[1] = *reinterpret_cast<const u32x4*>(Abuf + abase + kk + 16);  // K+16..23
#pragma unroll
      for (int g = 0; g < 7; ++g) {
        union { u32x4 q[2]; v16bf v; } bm;
        const unsigned short* wp = Wt + wtb[g] + kk;
        bm.q[0] = *reinterpret_cast<const u32x4*>(wp);
        bm.q[1] = *reinterpret_cast<const u32x4*>(wp + 16);
        acc[g] = __builtin_amdgcn_wmma_f32_16x16x32_bf16(
            false, a.v, false, bm.v, (short)0, acc[g], false, false);
      }
    }

    // ---- K-half handoff: s==1 publishes its 7 partial sums via LDS
    if (s) {
#pragma unroll
      for (int g = 0; g < 7; ++g) {
        union { v8f v; u32x4 q[2]; } p; p.v = acc[g];
        u32x4* d = reinterpret_cast<u32x4*>(scr + (size_t)(sbase + g * 32) * 8);
        d[0] = p.q[0]; d[1] = p.q[1];
      }
    }
    __syncthreads();

    // ---- s==0 waves: reduce K-halves, then full cell update in registers
    if (!s) {
#pragma unroll
      for (int g = 0; g < 7; ++g) {
        union { v8f v; u32x4 q[2]; } p;
        const u32x4* r = reinterpret_cast<const u32x4*>(scr + (size_t)(sbase + g * 32) * 8);
        p.q[0] = r[0]; p.q[1] = r[1];
        acc[g] += p.v;
      }
      float dts[8];
#pragma unroll
      for (int r = 0; r < 8; ++r) {
        int row = m * 16 + hi * 8 + r;
        dts[r] = dtime[row * T + t];
      }
#pragma unroll
      for (int r = 0; r < 8; ++r) {
        float gi  = sigf(acc[0][r] + bk[0]);
        float gf  = sigf(acc[1][r] + bk[1]);
        float go  = sigf(acc[2][r] + bk[2]);
        float gib = sigf(acc[3][r] + bk[3]);
        float gfb = sigf(acc[4][r] + bk[4]);
        float gd  = softplusf(acc[5][r] + bk[5]);
        float z   = tanhf(acc[6][r] + bk[6]);
        float c   = gf  * c_func[r] + gi  * z;
        float cb  = gfb * c_bar[r]  + gib * z;
        float cf  = cb + (c - cb) * __expf(-gd * dts[r]);
        float h   = go * tanhf(cf);
        c_func[r] = cf;
        c_bar[r]  = cb;
        int row = m * 16 + hi * 8 + r;
        hout[(size_t)row * T * H + (size_t)t * H + col] = h;
        float* dp = dout + (size_t)row * T * 4 * H + (size_t)t * 4 * H + col;
        dp[0]     = c;
        dp[H]     = cb;
        dp[2 * H] = gd;
        dp[3 * H] = go;
      }
    }

    // ---- Device-wide step barrier: h_t must be visible to all 32 blocks
    __threadfence();
    __syncthreads();
    if (tid == 0) {
      __hip_atomic_fetch_add(bar, 1u, __ATOMIC_ACQ_REL, __HIP_MEMORY_SCOPE_AGENT);
      unsigned tgt = (unsigned)NBLK * (unsigned)(t + 1);
      while (__hip_atomic_load(bar, __ATOMIC_ACQUIRE, __HIP_MEMORY_SCOPE_AGENT) < tgt) {
        __builtin_amdgcn_s_sleep(1);
      }
    }
    __syncthreads();
  }
}

// ---------------------------------------------------------------------------
extern "C" void kernel_launch(void* const* d_in, const int* in_sizes, int n_in,
                              void* d_out, int out_size, void* d_ws, size_t ws_size,
                              hipStream_t stream) {
  const float* seq   = (const float*)d_in[0];  // [B,T,H]
  const float* dtime = (const float*)d_in[1];  // [B,T]
  const float* W     = (const float*)d_in[2];  // [2H,7H]
  const float* bias  = (const float*)d_in[3];  // [7H]

  float* hout = (float*)d_out;                       // [B,T,H]
  float* dout = hout + (size_t)Bb * T * H;           // [B,T,4,H]

  // Workspace layout
  uint8_t* ws = (uint8_t*)d_ws;
  unsigned short* Wt = (unsigned short*)ws;                       // 7,340,032 B
  size_t off = (size_t)G7 * K2H * sizeof(unsigned short);
  float* hinit = (float*)(ws + off);                              // 131,072 B
  off += (size_t)Bb * H * sizeof(float);
  unsigned* bar = (unsigned*)(ws + off);

  ctl_init<<<(Bb * H + 255) / 256, 256, 0, stream>>>(hinit, bar);
  ctl_wconvert<<<(K2H * G7 + 255) / 256, 256, 0, stream>>>(W, Wt);

  const int ldsBytes = 64 * AROW * (int)sizeof(unsigned short)    // 132,096 B A-buffer
                     + 4 * 7 * 32 * 8 * (int)sizeof(float);       //  28,672 B K-half scratch
  (void)hipFuncSetAttribute((const void*)ctl_scan,
                            hipFuncAttributeMaxDynamicSharedMemorySize, ldsBytes);
  ctl_scan<<<NBLK, NTHR, ldsBytes, stream>>>(seq, dtime, Wt, bias, hinit,
                                             hout, dout, bar);
}
